// GraphEncoder_85298050499089
// MI455X (gfx1250) — matrix operand order repaired and checked by reading, not compiled, not run on hardware
//
#include <hip/hip_runtime.h>
#include <hip/hip_fp16.h>

// ---------------------------------------------------------------------------
// GAT graph encoder for MI455X (gfx1250, wave32, WMMA).
// GEMMs: v_wmma_f32_16x16x32_f16 (f16 operands, f32 accumulate).
//  - ep [E+N,128] never materialized: a_edge = eaA @ (lin_edge_w ⋈ att_edge).
//  - eaA stored once in f16 (WMMA A operand) to halve HBM traffic.
//  - Weights pre-packed into WMMA B-fragment layout: each lane's 16 halves are
//    contiguous -> 2x global_load_b128 per fragment (no scalar u16 + v_mov_b16).
//  - A tiles staged fp32->f16 through LDS with float4 loads / b64 stores,
//    fragments read via ds_load_b128.
// ---------------------------------------------------------------------------

typedef __attribute__((ext_vector_type(16))) _Float16 v16h;
typedef __attribute__((ext_vector_type(8)))  _Float16 v8h;
typedef __attribute__((ext_vector_type(4)))  _Float16 v4h;
typedef __attribute__((ext_vector_type(8)))  float    v8f;

#define NN 50000
#define NE 600000
#define EA (NE + NN)   // edges + self loops
#define HID 128

union AF { v16h v; v8h p[2]; _Float16 e[16]; };

// ---- fragment loaders (CDNA5 wave32 WMMA layouts, CDNA5 ISA) ----

// A 16x32 f16 fragment from an LDS tile (row pitch in halves, 16B aligned runs)
__device__ __forceinline__ v16h frag_a_lds(const _Float16* As, int pitch, int kb, int lane) {
  int m = lane & 15, g = lane >> 4;
  AF f;
  f.p[0] = *reinterpret_cast<const v8h*>(As + m * pitch + kb + 8 * g);        // K = kb+8g .. +7
  f.p[1] = *reinterpret_cast<const v8h*>(As + m * pitch + kb + 16 + 8 * g);   // K = kb+16+8g..+7
  return f.v;
}

// B 32x16 f16 fragment from pre-packed weights: lane's 16 halves are contiguous
__device__ __forceinline__ v16h frag_b_pk(const _Float16* Wp, int K, int kb, int n0, int lane) {
  const _Float16* p = Wp + ((((n0 >> 4) * (K >> 5)) + (kb >> 5)) * 32 + lane) * 16;
  AF f;
  f.p[0] = *reinterpret_cast<const v8h*>(p);
  f.p[1] = *reinterpret_cast<const v8h*>(p + 8);
  return f.v;
}

// ---- block GEMM: 16 rows x 128 cols, A fp32 global (lda=K) staged->f16 LDS,
//      8 waves each own one 16-col tile. Result left in LDS Cs[16][128]. ----
template <int K>
__device__ __forceinline__ void gemm16xK(const float* A, long long row0, long long M,
                                         const _Float16* Wp, float ascale,
                                         _Float16* As, float* Cs) {
  const int tid = threadIdx.x;
  // stage A tile: float4 global loads, v4h LDS stores
  for (int idx = tid; idx < 16 * (K / 4); idx += 256) {
    int r = idx / (K / 4), c = (idx - r * (K / 4)) * 4;
    long long row = row0 + r;
    float4 v = {0.f, 0.f, 0.f, 0.f};
    if (row < M) v = *reinterpret_cast<const float4*>(A + row * (long long)K + c);
    v4h hv = {(_Float16)(v.x * ascale), (_Float16)(v.y * ascale),
              (_Float16)(v.z * ascale), (_Float16)(v.w * ascale)};
    *reinterpret_cast<v4h*>(As + r * K + c) = hv;
  }
  __syncthreads();
  const int wave = tid >> 5, lane = tid & 31;
  v8f acc = {0.f, 0.f, 0.f, 0.f, 0.f, 0.f, 0.f, 0.f};
#pragma unroll
  for (int kb = 0; kb < K; kb += 32) {
    v16h a = frag_a_lds(As, K, kb, lane);
    v16h b = frag_b_pk(Wp, K, kb, wave * 16, lane);
    acc = __builtin_amdgcn_wmma_f32_16x16x32_f16(false, a, false, b, (short)0, acc, false, false);
  }
  int n = wave * 16 + (lane & 15), g = lane >> 4;
#pragma unroll
  for (int r = 0; r < 8; ++r) Cs[(r + 8 * g) * HID + n] = acc[r];
  __syncthreads();
}

// per-row LayerNorm stats over 128 cols held by 16 threads (8 vals each)
__device__ __forceinline__ void ln_stats(const float v[8], float& mu, float& rstd) {
  float s = 0.f, s2 = 0.f;
#pragma unroll
  for (int j = 0; j < 8; ++j) { s += v[j]; s2 += v[j] * v[j]; }
#pragma unroll
  for (int m = 8; m >= 1; m >>= 1) { s += __shfl_xor(s, m, 16); s2 += __shfl_xor(s2, m, 16); }
  mu = s * (1.0f / 128.0f);
  float var = s2 * (1.0f / 128.0f) - mu * mu;
  rstd = rsqrtf(var + 1e-5f);
}

// ---- ordered-int float max encoding ----
__device__ __forceinline__ unsigned fkey(float x) {
  unsigned u = __float_as_uint(x);
  return (u >> 31) ? ~u : (u | 0x80000000u);
}
__device__ __forceinline__ float funkey(unsigned k) {
  unsigned u = (k & 0x80000000u) ? (k & 0x7FFFFFFFu) : ~k;
  return __uint_as_float(u);
}

// ===========================================================================
// Kernels
// ===========================================================================

__global__ void k_fill_f32(float* p, float v, long long n) {
  long long t = (long long)blockIdx.x * blockDim.x + threadIdx.x;
  if (t < n) p[t] = v;
}
__global__ void k_fill_u32(unsigned* p, unsigned v, long long n) {
  long long t = (long long)blockIdx.x * blockDim.x + threadIdx.x;
  if (t < n) p[t] = v;
}

// Pack W[K][ncols] (f32, row-major) into WMMA B-fragment order (f16):
// flat t = (((ntile*(K/32) + kb)*32 + lane)*16 + h)
//   -> W[(kb*32 + 8*(lane>>4) + (h&7) + 16*(h>>3)) * ncols + ntile*16 + (lane&15)]
__global__ void k_pack_b(const float* __restrict__ W, _Float16* __restrict__ Wp,
                         int K, int ncols) {
  int t = blockIdx.x * blockDim.x + threadIdx.x;
  if (t >= K * ncols) return;
  int h = t & 15;
  int lane = (t >> 4) & 31;
  int rest = t >> 9;
  int kb = rest % (K >> 5);
  int ntile = rest / (K >> 5);
  int k = kb * 32 + 8 * (lane >> 4) + (h & 7) + 16 * (h >> 3);
  int n = ntile * 16 + (lane & 15);
  Wp[t] = (_Float16)W[k * ncols + n];
}

// wedge[k][h] = sum_d lin_edge_w[k][h*16+d] * att_edge[h][d]   ([128][16] f32, h>=8 -> 0)
__global__ void k_build_wedge(const float* __restrict__ lew, const float* __restrict__ atte,
                              float* __restrict__ wedge_f) {
  int t = blockIdx.x * blockDim.x + threadIdx.x;
  if (t >= 128 * 16) return;
  int k = t >> 4, h = t & 15;
  float s = 0.f;
  if (h < 8) {
#pragma unroll
    for (int d = 0; d < 16; ++d) s += lew[k * 128 + h * 16 + d] * atte[h * 16 + d];
  }
  wedge_f[k * 16 + h] = s;
}

// node projection: x = relu(LN(nf @ Wn + b)) -> xbuf (f32)
__global__ void __launch_bounds__(256) k_node_proj(const float* __restrict__ nf,
    const _Float16* __restrict__ Wn, const float* __restrict__ bias,
    const float* __restrict__ gam, const float* __restrict__ bet, float* __restrict__ xout) {
  __shared__ __align__(16) _Float16 As[16 * 256];
  __shared__ float Cs[16 * 128];
  long long row0 = (long long)blockIdx.x * 16;
  gemm16xK<256>(nf, row0, NN, Wn, 1.0f, As, Cs);
  int r = threadIdx.x >> 4, lg = threadIdx.x & 15;
  float v[8];
#pragma unroll
  for (int j = 0; j < 8; ++j) { int c = lg + j * 16; v[j] = Cs[r * 128 + c] + bias[c]; }
  float mu, rstd; ln_stats(v, mu, rstd);
  long long row = row0 + r;
  if (row < NN) {
#pragma unroll
    for (int j = 0; j < 8; ++j) {
      int c = lg + j * 16;
      float y = fmaxf((v[j] - mu) * rstd * gam[c] + bet[c], 0.0f);
      xout[row * 128 + c] = y;
    }
  }
}

// edge projection: ea = relu(LN(ef @ We + b)); write f16 row into eaA[e],
// accumulate per-dst sums (for mean self-loop attr) + in-degree counts.
__global__ void __launch_bounds__(256) k_edge_proj(const float* __restrict__ ef,
    const _Float16* __restrict__ We, const float* __restrict__ bias,
    const float* __restrict__ gam, const float* __restrict__ bet,
    const int* __restrict__ ei, _Float16* __restrict__ eaA,
    float* __restrict__ loop_acc, float* __restrict__ cnt) {
  __shared__ __align__(16) _Float16 As[16 * 64];
  __shared__ float Cs[16 * 128];
  long long row0 = (long long)blockIdx.x * 16;
  gemm16xK<64>(ef, row0, NE, We, 1.0f, As, Cs);
  int r = threadIdx.x >> 4, lg = threadIdx.x & 15;
  float v[8];
#pragma unroll
  for (int j = 0; j < 8; ++j) { int c = lg + j * 16; v[j] = Cs[r * 128 + c] + bias[c]; }
  float mu, rstd; ln_stats(v, mu, rstd);
  long long e = row0 + r;
  if (e < NE) {
    int d = ei[NE + e];  // edge_index[1] = dst
    if (lg == 0) atomicAdd(&cnt[d], 1.0f);
#pragma unroll
    for (int j = 0; j < 8; ++j) {
      int c = lg + j * 16;
      float y = fmaxf((v[j] - mu) * rstd * gam[c] + bet[c], 0.0f);
      eaA[e * 128 + c] = (_Float16)y;
      atomicAdd(&loop_acc[(long long)d * 128 + c], y);
    }
  }
}

// self-loop rows of eaA: mean of incoming edge attrs
__global__ void k_looprow(const float* __restrict__ loop_acc, const float* __restrict__ cnt,
                          _Float16* __restrict__ eaA) {
  long long t = (long long)blockIdx.x * blockDim.x + threadIdx.x;
  if (t >= (long long)NN * 128) return;
  long long n = t >> 7;
  eaA[(long long)NE * 128 + t] = (_Float16)(loop_acc[t] / fmaxf(cnt[n], 1.0f));
}

// xp = (ascale * x) @ lin_w ; also fold a_src/a_dst head reductions in epilogue
__global__ void __launch_bounds__(256) k_xp(const float* __restrict__ x,
    const _Float16* __restrict__ Lw, float ascale,
    const float* __restrict__ atts, const float* __restrict__ attd,
    float* __restrict__ xp, float* __restrict__ a_src, float* __restrict__ a_dst) {
  __shared__ __align__(16) _Float16 As[16 * 128];
  __shared__ float Cs[16 * 128];
  long long row0 = (long long)blockIdx.x * 16;
  gemm16xK<128>(x, row0, NN, Lw, ascale, As, Cs);
  int tid = threadIdx.x;
  {
    int r = tid >> 4, lg = tid & 15;
    long long row = row0 + r;
    if (row < NN) {
#pragma unroll
      for (int j = 0; j < 8; ++j) { int c = lg + j * 16; xp[row * 128 + c] = Cs[r * 128 + c]; }
    }
  }
  if (tid < 128) {
    int r = tid >> 3, h = tid & 7;
    long long row = row0 + r;
    if (row < NN) {
      float ss = 0.f, sd = 0.f;
#pragma unroll
      for (int d = 0; d < 16; ++d) {
        float xv = Cs[r * 128 + h * 16 + d];
        ss += xv * atts[h * 16 + d];
        sd += xv * attd[h * 16 + d];
      }
      a_src[row * 8 + h] = ss;
      a_dst[row * 8 + h] = sd;
    }
  }
}

// a_edge = eaA(f16) @ wedge(packed f16, 8 valid cols); pure WMMA, no staging
__global__ void __launch_bounds__(256) k_aedge(const _Float16* __restrict__ eaA,
                                               const _Float16* __restrict__ wedge,
                                               float* __restrict__ aeb) {
  int tid = threadIdx.x, wave = tid >> 5, lane = tid & 31;
  long long row0 = ((long long)blockIdx.x * 8 + wave) * 16;
  if (row0 >= EA) return;  // wave-uniform: EXEC stays all-ones for WMMA
  int m = lane & 15, g = lane >> 4;
  long long arow = row0 + m;
  bool ok = arow < EA;
  const _Float16* Ap = eaA + arow * 128;
  v8f acc = {0.f, 0.f, 0.f, 0.f, 0.f, 0.f, 0.f, 0.f};
  v8h zero = {};
#pragma unroll
  for (int kb = 0; kb < 128; kb += 32) {
    AF a;
    a.p[0] = ok ? *reinterpret_cast<const v8h*>(Ap + kb + 8 * g) : zero;
    a.p[1] = ok ? *reinterpret_cast<const v8h*>(Ap + kb + 16 + 8 * g) : zero;
    v16h b = frag_b_pk(wedge, 128, kb, 0, lane);
    acc = __builtin_amdgcn_wmma_f32_16x16x32_f16(false, a.v, false, b, (short)0, acc, false, false);
  }
  int n = lane & 15;
  if (n < 8) {
#pragma unroll
    for (int r = 0; r < 8; ++r) {
      long long row = row0 + r + 8 * g;
      if (row < EA) aeb[row * 8 + n] = acc[r];
    }
  }
}

// alpha = leaky_relu(a_src[src]+a_dst[dst]+a_edge); segment max via ordered-int atomicMax
__global__ void k_alpha(const int* __restrict__ ei, const float* __restrict__ a_src,
                        const float* __restrict__ a_dst, float* __restrict__ aeb,
                        unsigned* __restrict__ amax) {
  long long t = (long long)blockIdx.x * blockDim.x + threadIdx.x;
  if (t >= (long long)EA * 8) return;
  long long e = t >> 3; int h = (int)(t & 7);
  int s, d;
  if (e < NE) { s = ei[e]; d = ei[NE + e]; } else { s = d = (int)(e - NE); }
  float a = a_src[(long long)s * 8 + h] + a_dst[(long long)d * 8 + h] + aeb[t];
  a = (a > 0.f) ? a : 0.2f * a;   // PyG leaky_relu slope
  aeb[t] = a;
  atomicMax(&amax[(long long)d * 8 + h], fkey(a));
}

// ex = exp(alpha - amax[dst]); segment sum -> denom
__global__ void k_expsum(const int* __restrict__ ei, const unsigned* __restrict__ amax,
                         float* __restrict__ aeb, float* __restrict__ denom) {
  long long t = (long long)blockIdx.x * blockDim.x + threadIdx.x;
  if (t >= (long long)EA * 8) return;
  long long e = t >> 3; int h = (int)(t & 7);
  int d = (e < NE) ? ei[NE + e] : (int)(e - NE);
  float m = funkey(amax[(long long)d * 8 + h]);
  float ex = __expf(aeb[t] - m);
  aeb[t] = ex;
  atomicAdd(&denom[(long long)d * 8 + h], ex);
}

// scatter messages: out[dst] += xp[src] * softmax_w   (128 lanes per edge)
__global__ void __launch_bounds__(256) k_msg(const int* __restrict__ ei,
    const float* __restrict__ aeb, const float* __restrict__ denom,
    const float* __restrict__ xp, float* __restrict__ outb) {
  long long t = (long long)blockIdx.x * 256 + threadIdx.x;
  long long e = t >> 7; int c = (int)(t & 127);
  if (e >= EA) return;
  int s, d;
  if (e < NE) { s = ei[e]; d = ei[NE + e]; } else { s = d = (int)(e - NE); }
  int h = c >> 4;
  float w = aeb[e * 8 + h] / (denom[(long long)d * 8 + h] + 1e-16f);
  atomicAdd(&outb[(long long)d * 128 + c], xp[(long long)s * 128 + c] * w);
}

// finalize: y = LN(out + gat_bias); optional relu; optional graph-mean accumulation
__global__ void __launch_bounds__(256) k_finalize(const float* __restrict__ outb,
    const float* __restrict__ gbias, const float* __restrict__ gam,
    const float* __restrict__ bet, float* __restrict__ dst, int relu,
    float* __restrict__ gmean) {
  int wave = threadIdx.x >> 5, lane = threadIdx.x & 31;
  long long row = (long long)blockIdx.x * 8 + wave;
  if (row >= NN) return;
  float v[4]; float s = 0.f, s2 = 0.f;
#pragma unroll
  for (int j = 0; j < 4; ++j) {
    int c = lane + j * 32;
    float x = outb[row * 128 + c] + gbias[c];
    v[j] = x; s += x; s2 += x * x;
  }
#pragma unroll
  for (int m = 16; m >= 1; m >>= 1) { s += __shfl_xor(s, m, 32); s2 += __shfl_xor(s2, m, 32); }
  float mu = s * (1.0f / 128.0f);
  float var = s2 * (1.0f / 128.0f) - mu * mu;
  float rstd = rsqrtf(var + 1e-5f);
#pragma unroll
  for (int j = 0; j < 4; ++j) {
    int c = lane + j * 32;
    float y = (v[j] - mu) * rstd * gam[c] + bet[c];
    if (relu) y = fmaxf(y, 0.0f);
    dst[row * 128 + c] = y;
    if (gmean) atomicAdd(&gmean[c], y * (1.0f / (float)NN));
  }
}

// ===========================================================================
// Host orchestration (stream-only; graph-capture safe)
// ===========================================================================
extern "C" void kernel_launch(void* const* d_in, const int* in_sizes, int n_in,
                              void* d_out, int out_size, void* d_ws, size_t ws_size,
                              hipStream_t stream) {
  // Input order: top-level dict insertion order; nested 'params' pytree in
  // sorted-key (JAX tree) order:
  //  0 node_features [N,256]   1 edge_features [E,64]   2 edge_index [2,E] int32
  //  3 edge_proj.b  4 edge_proj.ln_b  5 edge_proj.ln_g  6 edge_proj.w [64,128]
  //  7..12  gat[0]: att_dst, att_edge, att_src, bias, lin_edge_w, lin_w
  //  13..18 gat[1]: att_dst, att_edge, att_src, bias, lin_edge_w, lin_w
  //  19 ln0.b  20 ln0.g  21 ln1.b  22 ln1.g
  //  23 node_proj.b  24 node_proj.ln_b  25 node_proj.ln_g  26 node_proj.w [256,128]
  const float* nf = (const float*)d_in[0];
  const float* ef = (const float*)d_in[1];
  const int*   ei = (const int*)d_in[2];
  const float* ep_b   = (const float*)d_in[3];
  const float* ep_lnb = (const float*)d_in[4];
  const float* ep_lng = (const float*)d_in[5];
  const float* ep_w   = (const float*)d_in[6];
  const float* g_attd[2] = {(const float*)d_in[7],  (const float*)d_in[13]};
  const float* g_atte[2] = {(const float*)d_in[8],  (const float*)d_in[14]};
  const float* g_atts[2] = {(const float*)d_in[9],  (const float*)d_in[15]};
  const float* g_bias[2] = {(const float*)d_in[10], (const float*)d_in[16]};
  const float* g_lew[2]  = {(const float*)d_in[11], (const float*)d_in[17]};
  const float* g_lw[2]   = {(const float*)d_in[12], (const float*)d_in[18]};
  const float* lnb[2] = {(const float*)d_in[19], (const float*)d_in[21]};
  const float* lng[2] = {(const float*)d_in[20], (const float*)d_in[22]};
  const float* np_b   = (const float*)d_in[23];
  const float* np_lnb = (const float*)d_in[24];
  const float* np_lng = (const float*)d_in[25];
  const float* np_w   = (const float*)d_in[26];

  // workspace carve-out (256B aligned)
  char* base = (char*)d_ws;
  size_t off = 0;
  auto carve = [&](size_t bytes) -> void* {
    void* p = base + off;
    off = (off + bytes + 255) & ~(size_t)255;
    return p;
  };
  float*     xbuf   = (float*)carve((size_t)NN * 128 * 4);
  float*     xp     = (float*)carve((size_t)NN * 128 * 4);
  float*     outb   = (float*)carve((size_t)NN * 128 * 4);   // also loop_acc
  float*     cnt    = (float*)carve((size_t)NN * 4);
  float*     a_src  = (float*)carve((size_t)NN * 8 * 4);
  float*     a_dst  = (float*)carve((size_t)NN * 8 * 4);
  unsigned*  amax   = (unsigned*)carve((size_t)NN * 8 * 4);
  float*     denom  = (float*)carve((size_t)NN * 8 * 4);
  float*     aeb    = (float*)carve((size_t)EA * 8 * 4);     // a_edge -> alpha -> ex
  _Float16*  eaA    = (_Float16*)carve((size_t)EA * 128 * 2);
  _Float16*  Wn_h   = (_Float16*)carve((size_t)256 * 128 * 2);  // packed B layout
  _Float16*  We_h   = (_Float16*)carve((size_t)64 * 128 * 2);   // packed B layout
  _Float16*  Lw_h[2]    = {(_Float16*)carve((size_t)128 * 128 * 2),
                           (_Float16*)carve((size_t)128 * 128 * 2)};
  _Float16*  wedge_h[2] = {(_Float16*)carve((size_t)128 * 16 * 2),
                           (_Float16*)carve((size_t)128 * 16 * 2)};
  float*     wedge_f = (float*)carve((size_t)128 * 16 * 4);
  (void)ws_size; (void)in_sizes; (void)n_in; (void)out_size;

  auto cdiv = [](long long a, long long b) { return (unsigned)((a + b - 1) / b); };

  // --- weight prep: fold lin_edge_w with att_edge; pack all B operands ---
  k_pack_b<<<cdiv(256 * 128, 256), 256, 0, stream>>>(np_w, Wn_h, 256, 128);
  k_pack_b<<<cdiv(64 * 128, 256), 256, 0, stream>>>(ep_w, We_h, 64, 128);
  for (int l = 0; l < 2; ++l) {
    k_pack_b<<<cdiv(128 * 128, 256), 256, 0, stream>>>(g_lw[l], Lw_h[l], 128, 128);
    k_build_wedge<<<cdiv(128 * 16, 256), 256, 0, stream>>>(g_lew[l], g_atte[l], wedge_f);
    k_pack_b<<<cdiv(128 * 16, 256), 256, 0, stream>>>(wedge_f, wedge_h[l], 128, 16);
  }

  // --- projections + self-loop attrs ---
  k_fill_f32<<<cdiv((long long)NN * 128, 256), 256, 0, stream>>>(outb, 0.f, (long long)NN * 128);
  k_fill_f32<<<cdiv(NN, 256), 256, 0, stream>>>(cnt, 0.f, NN);
  k_node_proj<<<cdiv(NN, 16), 256, 0, stream>>>(nf, Wn_h, np_b, np_lng, np_lnb, xbuf);
  k_edge_proj<<<cdiv(NE, 16), 256, 0, stream>>>(ef, We_h, ep_b, ep_lng, ep_lnb, ei, eaA, outb, cnt);
  k_looprow<<<cdiv((long long)NN * 128, 256), 256, 0, stream>>>(outb, cnt, eaA);

  // --- GAT layers ---
  for (int l = 0; l < 2; ++l) {
    float ascale = (l == 0) ? 1.0f : 2.0f;  // layer-1 residual: x = x + embeddings[-1] = 2x
    k_xp<<<cdiv(NN, 16), 256, 0, stream>>>(xbuf, Lw_h[l], ascale, g_atts[l], g_attd[l],
                                           xp, a_src, a_dst);
    k_aedge<<<cdiv(EA, 128), 256, 0, stream>>>(eaA, wedge_h[l], aeb);

    k_fill_u32<<<cdiv((long long)NN * 8, 256), 256, 0, stream>>>(amax, 0u, (long long)NN * 8);
    k_fill_f32<<<cdiv((long long)NN * 8, 256), 256, 0, stream>>>(denom, 0.f, (long long)NN * 8);
    k_fill_f32<<<cdiv((long long)NN * 128, 256), 256, 0, stream>>>(outb, 0.f, (long long)NN * 128);

    k_alpha<<<cdiv((long long)EA * 8, 256), 256, 0, stream>>>(ei, a_src, a_dst, aeb, amax);
    k_expsum<<<cdiv((long long)EA * 8, 256), 256, 0, stream>>>(ei, amax, aeb, denom);
    k_msg<<<cdiv((long long)EA * 128, 256), 256, 0, stream>>>(ei, aeb, denom, xp, outb);

    if (l == 0) {
      k_finalize<<<cdiv(NN, 8), 256, 0, stream>>>(outb, g_bias[0], lng[0], lnb[0],
                                                  xbuf, 1, nullptr);
    } else {
      float* out_x = (float*)d_out;
      float* gmean = out_x + (long long)NN * 128;
      k_fill_f32<<<1, 256, 0, stream>>>(gmean, 0.f, 128);
      k_finalize<<<cdiv(NN, 8), 256, 0, stream>>>(outb, g_bias[1], lng[1], lnb[1],
                                                  out_x, 0, gmean);
    }
  }
}
